// PyG_PointTransformerSegModel_6545530159672
// MI455X (gfx1250) — matrix operand order repaired and checked by reading, not compiled, not run on hardware
//
#include <hip/hip_runtime.h>
#include <hip/hip_bf16.h>

typedef __attribute__((ext_vector_type(16))) _Float16 v16h;
typedef __attribute__((ext_vector_type(8)))  _Float16 v8h;
typedef __attribute__((ext_vector_type(8)))  float    v8f;

#define MTOT 16384
#define NPTS 4096
#define BN_EPS 1e-5f
#define FLT_BIG 3.402823466e38f

// ---------------------------------------------------------------------------
// Embed linear 1: h1[M,64] = feat[M,6] @ w1[6,64] + b1.  K=6 -> plain VALU.
// ---------------------------------------------------------------------------
__global__ void embed1_kernel(const float* __restrict__ feat,
                              const float* __restrict__ w1,
                              const float* __restrict__ b1,
                              float* __restrict__ h1) {
    int t = blockIdx.x * blockDim.x + threadIdx.x;
    if (t >= MTOT * 64) return;
    int i = t >> 6, c = t & 63;
    const float* fr = feat + (size_t)i * 6;
    float acc = b1[c];
#pragma unroll
    for (int d = 0; d < 6; ++d) acc += fr[d] * w1[d * 64 + c];
    h1[t] = acc;
}

// ---------------------------------------------------------------------------
// BatchNorm statistics folded with affine params:
//   stats[c]     = sc = gamma*rsqrt(var+eps)
//   stats[128+c] = sh = beta - mean*sc
// One block per channel, deterministic tree reduction.
// ---------------------------------------------------------------------------
__global__ void stats_kernel(const float* __restrict__ X, int C,
                             const float* __restrict__ gamma,
                             const float* __restrict__ beta,
                             float* __restrict__ stats) {
    __shared__ float ss[256];
    __shared__ float sq[256];
    int c = blockIdx.x;
    float s = 0.f, q = 0.f;
    for (int r = threadIdx.x; r < MTOT; r += blockDim.x) {
        float v = X[(size_t)r * C + c];
        s += v;
        q += v * v;
    }
    ss[threadIdx.x] = s;
    sq[threadIdx.x] = q;
    __syncthreads();
    for (int st = 128; st > 0; st >>= 1) {
        if ((int)threadIdx.x < st) {
            ss[threadIdx.x] += ss[threadIdx.x + st];
            sq[threadIdx.x] += sq[threadIdx.x + st];
        }
        __syncthreads();
    }
    if (threadIdx.x == 0) {
        float mu  = ss[0] * (1.0f / (float)MTOT);
        float var = sq[0] * (1.0f / (float)MTOT) - mu * mu;
        float sc  = gamma[c] * rsqrtf(var + BN_EPS);
        stats[c]       = sc;
        stats[128 + c] = beta[c] - mu * sc;
    }
}

// ---------------------------------------------------------------------------
// Apply BN(sc,sh)+ReLU once and materialize activations as f16 row-major:
//   Ah[i,c] = (half) relu(X[i,c]*sc[c] + sh[c])
// 4 elements per thread (float4 in, 8-byte half4 out). C is a power of two.
// ---------------------------------------------------------------------------
__global__ void pack_a_kernel(const float* __restrict__ X,
                              const float* __restrict__ stats, int C,
                              _Float16* __restrict__ Ah) {
    int t = blockIdx.x * blockDim.x + threadIdx.x;
    size_t base = (size_t)t * 4;
    if (base >= (size_t)MTOT * C) return;
    int c0 = (int)(base & (size_t)(C - 1));
    float4 p = *reinterpret_cast<const float4*>(X + base);
    union { _Float16 h[4]; float2 f2; } u;
#pragma unroll
    for (int e = 0; e < 4; ++e) {
        int c = c0 + e;
        float v = fmaf((&p.x)[e], stats[c], stats[128 + c]);
        u.h[e] = (_Float16)(v > 0.f ? v : 0.f);
    }
    *reinterpret_cast<float2*>(Ah + base) = u.f2;
}

// ---------------------------------------------------------------------------
// Pack W[K,N] (f32, row-major) into WMMA B-fragment layout (f16):
// fragment (ntile, kc): lane holds 16 halfs = K rows kc*32 + h*8 + t (t<8),
// kc*32 + 16 + h*8 + (t-8) (t>=8), h = lane/16, of column ntile*16 + lane%16.
// Single small block per weight matrix.
// ---------------------------------------------------------------------------
__global__ void pack_w_kernel(const float* __restrict__ W, int Ncols,
                              int KCH, int NT, _Float16* __restrict__ Wp) {
    int id = threadIdx.x;
    int lane = id & 31;
    int rest = id >> 5;                 // rest = ntile*KCH + kc
    if (rest >= NT * KCH) return;
    int kc    = rest % KCH;
    int ntile = rest / KCH;
    int col   = ntile * 16 + (lane & 15);
    int colc  = (col < Ncols) ? col : (Ncols - 1);
    int bbase = (lane < 16) ? 0 : 16;
    v16h b;
#pragma unroll
    for (int t = 0; t < 16; ++t) {
        int kb = kc * 32 + (t < 8 ? (bbase >> 1) + t : 16 + (bbase >> 1) + (t - 8));
        b[t] = (_Float16)W[(size_t)kb * Ncols + colc];
    }
    *reinterpret_cast<v16h*>(Wp + (size_t)(rest * 32 + lane) * 16) = b;
}

// ---------------------------------------------------------------------------
// Pure WMMA GEMM on pre-packed f16 operands:
//   C[M,N] = Ah[M,K] @ W[K,N] (+ bias)
// Each wave: 4 consecutive M-tiles x 1 N-tile; B fragment reused 4x; K fully
// unrolled (KCH chunks). A fragment = two aligned 16B half-vector loads.
// ---------------------------------------------------------------------------
template <int KCH>
__global__ void wmma_gemm(const _Float16* __restrict__ Ah,
                          const _Float16* __restrict__ Wp,
                          const float* __restrict__ bias,
                          float* __restrict__ C, int Ncols) {
    const int Kdim  = KCH * 32;
    const int lane  = threadIdx.x;       // 0..31
    const int ntile = threadIdx.y;       // N tile handled by this wave
    const int col   = ntile * 16 + (lane & 15);
    const int abase = (lane < 16) ? 0 : 8;
    const int mrow  = blockIdx.x * 64 + (lane & 15);   // row within M-tile 0

    v8f acc[4] = {};

#pragma unroll
    for (int kc = 0; kc < KCH; ++kc) {
        const v16h b = *reinterpret_cast<const v16h*>(
            Wp + (size_t)((ntile * KCH + kc) * 32 + lane) * 16);
        const int kb = kc * 32 + abase;
#pragma unroll
        for (int m = 0; m < 4; ++m) {
            const _Float16* Ar = Ah + (size_t)(mrow + m * 16) * Kdim + kb;
            if (m == 0 && kc + 1 < KCH) __builtin_prefetch(Ar + 32, 0, 0);
            v8h lo = *reinterpret_cast<const v8h*>(Ar);        // k+0..7
            v8h hi = *reinterpret_cast<const v8h*>(Ar + 16);   // k+16..23
            v16h a = __builtin_shufflevector(lo, hi, 0, 1, 2, 3, 4, 5, 6, 7,
                                             8, 9, 10, 11, 12, 13, 14, 15);
            acc[m] = __builtin_amdgcn_wmma_f32_16x16x32_f16(
                false, a, false, b, (short)0, acc[m], false, false);
        }
    }

    if (col < Ncols) {
        const float bv = bias ? bias[col] : 0.f;
        const int rhalf = (lane < 16) ? 0 : 8;
#pragma unroll
        for (int m = 0; m < 4; ++m) {
            const int rbase = blockIdx.x * 64 + m * 16 + rhalf;
#pragma unroll
            for (int j = 0; j < 8; ++j)
                C[(size_t)(rbase + j) * Ncols + col] = acc[m][j] + bv;
        }
    }
}

// ---------------------------------------------------------------------------
// Brute-force kNN (k=16) within each 4096-point batch block, loop=False.
// One block (256 threads) per query point; per-thread sorted top-16, then
// 16 rounds of LDS tree-merge with lexicographic (d2, j) ordering.
// Also emits rel[i,k,:] = pos_i - pos_j.
// ---------------------------------------------------------------------------
__global__ void knn_kernel(const float* __restrict__ feat,
                           int* __restrict__ idx,
                           float* __restrict__ rel) {
    const int i  = blockIdx.x;
    const int t  = threadIdx.x;          // 0..255
    const int j0 = i & ~(NPTS - 1);      // batch block start

    __shared__ float sd[256 * 16];
    __shared__ int   si[256 * 16];
    __shared__ float rv[256];
    __shared__ int   rj[256];
    __shared__ int   rt[256];
    __shared__ int   chosen[16];

    const float px = feat[(size_t)i * 6 + 0];
    const float py = feat[(size_t)i * 6 + 1];
    const float pz = feat[(size_t)i * 6 + 2];

    float ld[16];
    int   li[16];
#pragma unroll
    for (int k = 0; k < 16; ++k) { ld[k] = FLT_BIG; li[k] = 0x7fffffff; }

    for (int j = j0 + t; j < j0 + NPTS; j += 256) {
        if (j == i) continue;
        float dx = px - feat[(size_t)j * 6 + 0];
        float dy = py - feat[(size_t)j * 6 + 1];
        float dz = pz - feat[(size_t)j * 6 + 2];
        float d2 = dx * dx + dy * dy + dz * dz;
        if (d2 < ld[15] || (d2 == ld[15] && j < li[15])) {
            int p = 15;
            while (p > 0 && (ld[p - 1] > d2 || (ld[p - 1] == d2 && li[p - 1] > j))) {
                ld[p] = ld[p - 1];
                li[p] = li[p - 1];
                --p;
            }
            ld[p] = d2;
            li[p] = j;
        }
    }
#pragma unroll
    for (int k = 0; k < 16; ++k) { sd[t * 16 + k] = ld[k]; si[t * 16 + k] = li[k]; }
    __syncthreads();

    int ptr = 0;
    for (int r = 0; r < 16; ++r) {
        rv[t] = (ptr < 16) ? sd[t * 16 + ptr] : FLT_BIG;
        rj[t] = (ptr < 16) ? si[t * 16 + ptr] : 0x7fffffff;
        rt[t] = t;
        __syncthreads();
        for (int s = 128; s > 0; s >>= 1) {
            if (t < s) {
                float v2 = rv[t + s];
                int   j2 = rj[t + s];
                if (v2 < rv[t] || (v2 == rv[t] && j2 < rj[t])) {
                    rv[t] = v2; rj[t] = j2; rt[t] = rt[t + s];
                }
            }
            __syncthreads();
        }
        if (t == rt[0]) ++ptr;
        if (t == 0) chosen[r] = rj[0];
        __syncthreads();
    }

    if (t < 16) idx[(size_t)i * 16 + t] = chosen[t];
    if (t < 48) {
        int k = t / 3, d = t % 3;
        int jg = chosen[k];
        rel[(size_t)i * 48 + t] = feat[(size_t)i * 6 + d] - feat[(size_t)jg * 6 + d];
    }
}

// ---------------------------------------------------------------------------
// Point-transformer attention (H=128, k=16):
//   delta = rel @ pos_w + pos_b
//   alpha = softmax_k(a_dst_i - a_src_j + delta)
//   out_i = sum_k alpha * (v_lin_j + delta)    (raw, pre-BN)
// One block per node, one thread per H channel; softmax is thread-local.
// ---------------------------------------------------------------------------
__global__ void attn_kernel(const float* __restrict__ adst,
                            const float* __restrict__ asrc,
                            const float* __restrict__ vlin,
                            const int*   __restrict__ idx,
                            const float* __restrict__ rel,
                            const float* __restrict__ pw,
                            const float* __restrict__ pb,
                            float* __restrict__ out) {
    const int i = blockIdx.x;
    const int h = threadIdx.x;  // 0..127
    __shared__ int   sj[16];
    __shared__ float sr[48];
    if (h < 16) sj[h] = idx[(size_t)i * 16 + h];
    if (h < 48) sr[h] = rel[(size_t)i * 48 + h];
    __syncthreads();

    const float w0 = pw[h], w1 = pw[128 + h], w2 = pw[256 + h], bb = pb[h];
    const float ai = adst[(size_t)i * 128 + h];

    float ek[16], dk[16];
    float m = -FLT_BIG;
#pragma unroll
    for (int k = 0; k < 16; ++k) {
        int jg  = sj[k];
        float d = bb + sr[k * 3] * w0 + sr[k * 3 + 1] * w1 + sr[k * 3 + 2] * w2;
        float e = ai - asrc[(size_t)jg * 128 + h] + d;
        ek[k] = e;
        dk[k] = d;
        m = fmaxf(m, e);
    }
    float s = 0.f, acc = 0.f;
#pragma unroll
    for (int k = 0; k < 16; ++k) {
        float p = __expf(ek[k] - m);
        s += p;
        acc += p * (vlin[(size_t)sj[k] * 128 + h] + dk[k]);
    }
    out[(size_t)i * 128 + h] = acc / s;
}

// ---------------------------------------------------------------------------
// Host orchestration
// ---------------------------------------------------------------------------
extern "C" void kernel_launch(void* const* d_in, const int* in_sizes, int n_in,
                              void* d_out, int out_size, void* d_ws, size_t ws_size,
                              hipStream_t stream) {
    (void)out_size; (void)ws_size;
    const float* in[29];
    for (int i = 0; i < n_in && i < 29; ++i) in[i] = (const float*)d_in[i];

    const float *feat, *e_w1, *e_b1, *e_g1, *e_be1, *e_w2, *e_b2, *g_emb, *be_emb;
    const float *w_lin[2], *w_src[2], *w_dst[2], *pos_w[2], *pos_b[2], *lg[2], *lbe[2];
    const float *d_w1, *d_b1, *d_g, *d_be, *d_w2, *d_b2;

    feat = in[0];
    const bool insertion = (n_in > 1 && in_sizes[1] == 6 * 64);  // e_w1 first?
    if (insertion) {
        e_w1 = in[1];  e_b1 = in[2];  e_g1 = in[3];  e_be1 = in[4];
        e_w2 = in[5];  e_b2 = in[6];  g_emb = in[7]; be_emb = in[8];
        for (int l = 0; l < 2; ++l) {
            int b = 9 + 7 * l;
            w_lin[l] = in[b + 0]; w_src[l] = in[b + 1]; w_dst[l] = in[b + 2];
            pos_w[l] = in[b + 3]; pos_b[l] = in[b + 4];
            lg[l]    = in[b + 5]; lbe[l]   = in[b + 6];
        }
        d_w1 = in[23]; d_b1 = in[24]; d_g = in[25]; d_be = in[26];
        d_w2 = in[27]; d_b2 = in[28];
    } else {
        // jax pytree flatten: dict keys alphabetical, layer dicts alphabetical
        be_emb = in[1];  d_b1 = in[2];  d_b2 = in[3];  d_be = in[4];  d_g = in[5];
        d_w1 = in[6];    d_w2 = in[7];  e_b1 = in[8];  e_b2 = in[9];
        e_be1 = in[10];  e_g1 = in[11]; e_w1 = in[12]; e_w2 = in[13]; g_emb = in[14];
        for (int l = 0; l < 2; ++l) {
            int b = 15 + 7 * l;  // be, g, pos_b, pos_w, w_dst, w_lin, w_src
            lbe[l]   = in[b + 0]; lg[l]    = in[b + 1];
            pos_b[l] = in[b + 2]; pos_w[l] = in[b + 3];
            w_dst[l] = in[b + 4]; w_lin[l] = in[b + 5]; w_src[l] = in[b + 6];
        }
    }

    // ---- workspace layout ----
    const size_t M = MTOT;
    float* ws    = (float*)d_ws;
    float* h1    = ws;                    // M*64   embed linear1 raw
    float* bufX  = h1   + M * 64;         // M*128
    float* bufY  = bufX + M * 128;        // M*128
    float* adst  = bufY + M * 128;        // M*128
    float* asrc  = adst + M * 128;        // M*128
    float* vlin  = asrc + M * 128;        // M*128
    float* rel   = vlin + M * 128;        // M*48
    int*   idx   = (int*)(rel + M * 48);  // M*16
    float* stats = (float*)(idx + M * 16);// 256 floats (sc[0:128], sh[128:256])
    _Float16* wp = (_Float16*)(stats + 256);
    // packed weights (halfs): region sizes = NT*KCH*512
    _Float16* wp_ew2  = wp;                    // NT=4, KCH=2 -> 4096
    _Float16* wp_d0   = wp_ew2 + 4096;         // NT=8, KCH=2 -> 8192
    _Float16* wp_s0   = wp_d0  + 8192;
    _Float16* wp_l0   = wp_s0  + 8192;
    _Float16* wp_d1   = wp_l0  + 8192;         // NT=8, KCH=4 -> 16384
    _Float16* wp_s1   = wp_d1  + 16384;
    _Float16* wp_l1   = wp_s1  + 16384;
    _Float16* wp_dw1  = wp_l1  + 16384;        // NT=8, KCH=4 -> 16384
    _Float16* wp_dw2  = wp_dw1 + 16384;        // NT=1, KCH=4 -> 2048
    _Float16* Ah      = wp_dw2 + 2048;         // M*128 halfs, per-stage f16 A

    const dim3 blk32x1(32, 1), blk32x4(32, 4), blk32x8(32, 8);
    const int  MB = MTOT / 64;            // 256 blocks (4 M-tiles per wave)
    const int  PA64  = (MTOT * 64 / 4 + 255) / 256;    // pack_a grids
    const int  PA128 = (MTOT * 128 / 4 + 255) / 256;

    // --- pack all weight matrices into B-fragment layout (tiny one-off) ---
    pack_w_kernel<<<1,  4 * 2 * 32, 0, stream>>>(e_w2,     64, 2, 4, wp_ew2);
    pack_w_kernel<<<1,  8 * 2 * 32, 0, stream>>>(w_dst[0], 128, 2, 8, wp_d0);
    pack_w_kernel<<<1,  8 * 2 * 32, 0, stream>>>(w_src[0], 128, 2, 8, wp_s0);
    pack_w_kernel<<<1,  8 * 2 * 32, 0, stream>>>(w_lin[0], 128, 2, 8, wp_l0);
    pack_w_kernel<<<1,  8 * 4 * 32, 0, stream>>>(w_dst[1], 128, 4, 8, wp_d1);
    pack_w_kernel<<<1,  8 * 4 * 32, 0, stream>>>(w_src[1], 128, 4, 8, wp_s1);
    pack_w_kernel<<<1,  8 * 4 * 32, 0, stream>>>(w_lin[1], 128, 4, 8, wp_l1);
    pack_w_kernel<<<1,  8 * 4 * 32, 0, stream>>>(d_w1,     128, 4, 8, wp_dw1);
    pack_w_kernel<<<1,  1 * 4 * 32, 0, stream>>>(d_w2,      13, 4, 1, wp_dw2);

    // --- embed MLP ---
    embed1_kernel<<<(MTOT * 64 + 255) / 256, 256, 0, stream>>>(feat, e_w1, e_b1, h1);
    stats_kernel<<<64, 256, 0, stream>>>(h1, 64, e_g1, e_be1, stats);
    pack_a_kernel<<<PA64, 256, 0, stream>>>(h1, stats, 64, Ah);
    wmma_gemm<2><<<MB, blk32x4, 0, stream>>>(Ah, wp_ew2, e_b2, bufX, 64);
    // kNN graph + relative positions (independent of embed values)
    knn_kernel<<<MTOT, 256, 0, stream>>>(feat, idx, rel);
    stats_kernel<<<64, 256, 0, stream>>>(bufX, 64, g_emb, be_emb, stats);
    pack_a_kernel<<<PA64, 256, 0, stream>>>(bufX, stats, 64, Ah);

    // --- transformer layer 1 (Cin=64) ---
    wmma_gemm<2><<<MB, blk32x8, 0, stream>>>(Ah, wp_d0, nullptr, adst, 128);
    wmma_gemm<2><<<MB, blk32x8, 0, stream>>>(Ah, wp_s0, nullptr, asrc, 128);
    wmma_gemm<2><<<MB, blk32x8, 0, stream>>>(Ah, wp_l0, nullptr, vlin, 128);
    attn_kernel<<<MTOT, 128, 0, stream>>>(adst, asrc, vlin, idx, rel,
                                          pos_w[0], pos_b[0], bufY);
    stats_kernel<<<128, 256, 0, stream>>>(bufY, 128, lg[0], lbe[0], stats);
    pack_a_kernel<<<PA128, 256, 0, stream>>>(bufY, stats, 128, Ah);

    // --- transformer layer 2 (Cin=128) ---
    wmma_gemm<4><<<MB, blk32x8, 0, stream>>>(Ah, wp_d1, nullptr, adst, 128);
    wmma_gemm<4><<<MB, blk32x8, 0, stream>>>(Ah, wp_s1, nullptr, asrc, 128);
    wmma_gemm<4><<<MB, blk32x8, 0, stream>>>(Ah, wp_l1, nullptr, vlin, 128);
    attn_kernel<<<MTOT, 128, 0, stream>>>(adst, asrc, vlin, idx, rel,
                                          pos_w[1], pos_b[1], bufX);
    stats_kernel<<<128, 256, 0, stream>>>(bufX, 128, lg[1], lbe[1], stats);
    pack_a_kernel<<<PA128, 256, 0, stream>>>(bufX, stats, 128, Ah);

    // --- decoder ---
    wmma_gemm<4><<<MB, blk32x8, 0, stream>>>(Ah, wp_dw1, d_b1, bufY, 128);
    stats_kernel<<<128, 256, 0, stream>>>(bufY, 128, d_g, d_be, stats);
    pack_a_kernel<<<PA128, 256, 0, stream>>>(bufY, stats, 128, Ah);
    wmma_gemm<4><<<MB, blk32x1, 0, stream>>>(Ah, wp_dw2, d_b2, (float*)d_out, 13);
}
